// ProteinFeatures_19842748907686
// MI455X (gfx1250) — matrix operand order repaired and checked by reading, not compiled, not run on hardware
//
#include <hip/hip_runtime.h>
#include <hip/hip_bf16.h>

typedef __attribute__((ext_vector_type(16))) _Float16 v16h;
typedef __attribute__((ext_vector_type(8)))  _Float16 v8h;
typedef __attribute__((ext_vector_type(8)))  float    v8f;

#define PF_B 8
#define PF_L 2048
#define PF_K 30
#define PF_EDGES (PF_B * PF_L * PF_K)          // 491520
#define PF_EOUT  ((size_t)PF_EDGES * 128)      // 62914560 floats of E
#define PF_KDIM  416                            // 16 pos + 25*16 rbf = 13 * 32
#define PF_ROWP  448                            // padded LDS row (halves): 896B, 32B aligned
#define PF_WAVES 4

// Map an 8-half feature group index (0..51) to its swizzled position in the row.
// Within each 32-K block, group order becomes [K0..7, K16..23, K8..15, K24..31]
// so lanes 0-15 / 16-31 each read one contiguous 32B A-fragment per K-step.
__device__ __forceinline__ int pf_swz_group(int gi) {
    int blk = gi >> 2, sub = gi & 3;
    int subp = (sub == 1) ? 2 : (sub == 2) ? 1 : sub;   // swap groups 1<->2
    return (blk * 4 + subp) * 8;                         // half offset in row
}

// ---------------- Kernel 1: virtual C-beta ----------------
__global__ __launch_bounds__(256) void pf_cb_kernel(const float* __restrict__ X,
                                                    float* __restrict__ Cb, int n) {
    int i = blockIdx.x * blockDim.x + threadIdx.x;
    if (i >= n) return;
    size_t p = (size_t)i * 12;                    // 4 atoms * 3
    float Nx = X[p+0], Ny = X[p+1], Nz = X[p+2];
    float Ax = X[p+3], Ay = X[p+4], Az = X[p+5];  // Ca
    float Cx = X[p+6], Cy = X[p+7], Cz = X[p+8];
    float bx = Ax - Nx, by = Ay - Ny, bz = Az - Nz;
    float cx = Cx - Ax, cy = Cy - Ay, cz = Cz - Az;
    float ax = by*cz - bz*cy, ay = bz*cx - bx*cz, az = bx*cy - by*cx;
    Cb[(size_t)i*3+0] = -0.58273431f*ax + 0.56802827f*bx - 0.54067466f*cx + Ax;
    Cb[(size_t)i*3+1] = -0.58273431f*ay + 0.56802827f*by - 0.54067466f*cy + Ay;
    Cb[(size_t)i*3+2] = -0.58273431f*az + 0.56802827f*bz - 0.54067466f*cz + Az;
}

// ---------------- Kernel 2: W_e (416x128 f32) -> Wt (128x416 f16, K-contiguous) ----------------
__global__ __launch_bounds__(256) void pf_wt_kernel(const float* __restrict__ We,
                                                    _Float16* __restrict__ Wt, int n) {
    int id = blockIdx.x * blockDim.x + threadIdx.x;
    if (id >= n) return;
    int nn = id / PF_KDIM;
    int k  = id - nn * PF_KDIM;
    Wt[id] = (_Float16)We[(size_t)k * 128 + nn];
}

// ---------------- Kernel 3: masked top-30 nearest Ca neighbors ----------------
__global__ __launch_bounds__(256) void pf_topk_kernel(const float* __restrict__ X,
                                                      const float* __restrict__ mask,
                                                      int* __restrict__ Eidx) {
    __shared__ float sD[PF_L];
    __shared__ unsigned long long skey[256];
    __shared__ float smax[256];
    int tid = threadIdx.x;
    int bi  = blockIdx.x;
    int bb  = bi >> 11;
    int ii  = bi & (PF_L - 1);
    int node_i = bb * PF_L + ii;
    size_t pi = ((size_t)node_i * 4 + 1) * 3;     // Ca of residue i
    float cx = X[pi], cy = X[pi+1], cz = X[pi+2];
    float mi = mask[node_i];

    float lmax = 0.0f;
    for (int j = tid; j < PF_L; j += 256) {
        size_t pj = ((size_t)(bb * PF_L + j) * 4 + 1) * 3;
        float dx = cx - X[pj], dy = cy - X[pj+1], dz = cz - X[pj+2];
        float d = mi * mask[bb * PF_L + j] * sqrtf(dx*dx + dy*dy + dz*dz + 1e-6f);
        sD[j] = d;
        lmax = fmaxf(lmax, d);
    }
    smax[tid] = lmax;
    __syncthreads();
    for (int s = 128; s > 0; s >>= 1) {
        if (tid < s) smax[tid] = fmaxf(smax[tid], smax[tid + s]);
        __syncthreads();
    }
    float Dmax = smax[0];
    __syncthreads();
    for (int j = tid; j < PF_L; j += 256) {
        float m2 = mi * mask[bb * PF_L + j];
        sD[j] = sD[j] + (1.0f - m2) * Dmax;       // D_adjust
    }
    __syncthreads();

    for (int t = 0; t < PF_K; ++t) {
        unsigned long long best = ~0ull;
        for (int j = tid; j < PF_L; j += 256) {
            unsigned long long key =
                ((unsigned long long)__float_as_uint(sD[j]) << 32) | (unsigned int)j;
            best = key < best ? key : best;
        }
        skey[tid] = best;
        __syncthreads();
        for (int s = 128; s > 0; s >>= 1) {
            if (tid < s) {
                unsigned long long o = skey[tid + s];
                if (o < skey[tid]) skey[tid] = o;
            }
            __syncthreads();
        }
        unsigned long long w = skey[0];
        int jsel = (int)(w & 0xffffffffu);
        if (tid == 0) {
            Eidx[(size_t)node_i * PF_K + t] = jsel;
            sD[jsel] = __uint_as_float(0x7f7fffffu); // FLT_MAX: exclude
        }
        __syncthreads();
    }
}

// atom order: 0=Ca,1=N,2=C,3=O,4=Cb
__device__ __forceinline__ void pf_fetch(const float* __restrict__ X,
                                         const float* __restrict__ Cb,
                                         int node, int a,
                                         float& x, float& y, float& z) {
    if (a == 4) {
        size_t p = (size_t)node * 3;
        x = Cb[p]; y = Cb[p+1]; z = Cb[p+2];
    } else {
        int slot = (a == 0) ? 1 : (a == 1) ? 0 : (a == 2) ? 2 : 3;
        size_t p = ((size_t)node * 4 + slot) * 3;
        x = X[p]; y = X[p+1]; z = X[p+2];
    }
}

// ---------------- Kernel 4: feature build + WMMA GEMM + bias + LayerNorm ----------------
__global__ __launch_bounds__(128) void pf_edge_gemm_kernel(
    const float* __restrict__ X, const float* __restrict__ Cb,
    const int* __restrict__ residx, const int* __restrict__ chains,
    const _Float16* __restrict__ Wt,
    const float* __restrict__ b_e, const float* __restrict__ g_e,
    const float* __restrict__ b_ln,
    const float* __restrict__ W_pe, const float* __restrict__ b_pe,
    const int* __restrict__ Eidx, float* __restrict__ out) {

    __shared__ _Float16 sfeat[PF_WAVES][16][PF_ROWP];   // 4*16*448*2 = 57344 B

    const int lane = threadIdx.x & 31;
    const int wv   = threadIdx.x >> 5;
    const int tile = blockIdx.x * PF_WAVES + wv;
    const int e0   = tile * 16;

    // ---- build 16 x 416 feature tile in LDS (f16), swizzled for direct A-frag loads ----
    for (int it = 0; it < 13; ++it) {
        int item = lane + 32 * it;                 // 16 edges * 25 atom pairs
        if (item < 400) {
            int r = item / 25, p = item - r * 25;
            int e  = e0 + r;
            int bb = e / (PF_L * PF_K);
            int rem = e - bb * (PF_L * PF_K);
            int ii = rem / PF_K;
            int jj = Eidx[e];
            int ai = p / 5, aj = p - ai * 5;
            float ax, ay, az, bx, by, bz;
            pf_fetch(X, Cb, bb * PF_L + ii, ai, ax, ay, az);
            pf_fetch(X, Cb, bb * PF_L + jj, aj, bx, by, bz);
            float dx = ax - bx, dy = ay - by, dz = az - bz;
            float D = sqrtf(dx*dx + dy*dy + dz*dz + 1e-6f);
            v8h h0, h1;
            #pragma unroll
            for (int t = 0; t < 8; ++t) {
                float u = (D - (2.0f + (float)t * (20.0f / 15.0f))) * 0.8f;   // /1.25
                h0[t] = (_Float16)__expf(-u * u);
            }
            #pragma unroll
            for (int t = 0; t < 8; ++t) {
                float u = (D - (2.0f + (float)(t + 8) * (20.0f / 15.0f))) * 0.8f;
                h1[t] = (_Float16)__expf(-u * u);
            }
            int gi0 = 2 + 2 * p;                   // half-groups of this pair's 16 RBFs
            *(v8h*)&sfeat[wv][r][pf_swz_group(gi0)]     = h0;
            *(v8h*)&sfeat[wv][r][pf_swz_group(gi0 + 1)] = h1;
        }
    }
    if (lane < 16) {                               // positional embedding -> feature[0:16]
        int r = lane;
        int e  = e0 + r;
        int bb = e / (PF_L * PF_K);
        int rem = e - bb * (PF_L * PF_K);
        int ii = rem / PF_K;
        int jj = Eidx[e];
        int ni = bb * PF_L + ii, nj = bb * PF_L + jj;
        int off = residx[ni] - residx[nj];
        int same = (chains[ni] == chains[nj]) ? 1 : 0;
        int dcl = off + 32;
        dcl = dcl < 0 ? 0 : (dcl > 64 ? 64 : dcl);
        int d = same ? dcl : 65;
        v8h h0, h1;
        #pragma unroll
        for (int t = 0; t < 8; ++t) {
            h0[t] = (_Float16)(W_pe[d * 16 + t] + b_pe[t]);
            h1[t] = (_Float16)(W_pe[d * 16 + 8 + t] + b_pe[8 + t]);
        }
        *(v8h*)&sfeat[wv][r][pf_swz_group(0)] = h0;   // pos 0  (K 0..7)
        *(v8h*)&sfeat[wv][r][pf_swz_group(1)] = h1;   // pos 16 (K 8..15)
    }
    __syncthreads();

    // ---- WMMA: [16 x 416] @ [416 x 128], 13 K-steps x 8 N-tiles, fully unrolled ----
    const int hi = lane >> 4;       // lane half
    const int ln = lane & 15;
    v8f acc[8];
    #pragma unroll
    for (int nt = 0; nt < 8; ++nt) acc[nt] = (v8f){0,0,0,0,0,0,0,0};

    const _Float16* arow = &sfeat[wv][ln][hi * 16];             // swizzled A fragments
    const _Float16* wbase = Wt + (size_t)ln * PF_KDIM + hi * 8;  // B: col ln of each tile

    #pragma unroll
    for (int kk = 0; kk < 13; ++kk) {
        v16h a = *(const v16h*)(arow + kk * 32);                // one 32B LDS read
        #pragma unroll
        for (int nt = 0; nt < 8; ++nt) {
            v8h blo = *(const v8h*)(wbase + nt * 16 * PF_KDIM + kk * 32);
            v8h bhi = *(const v8h*)(wbase + nt * 16 * PF_KDIM + kk * 32 + 16);
            v16h bfr = __builtin_shufflevector(blo, bhi, 0, 1, 2, 3, 4, 5, 6, 7, 8, 9, 10, 11, 12, 13, 14, 15);
            acc[nt] = __builtin_amdgcn_wmma_f32_16x16x32_f16(
                false, a, false, bfr, (short)0, acc[nt], false, false);
        }
    }

    // ---- epilogue: +bias, LayerNorm over 128, *gamma + beta ----
    float bias[8], gg[8], bl[8];
    #pragma unroll
    for (int nt = 0; nt < 8; ++nt) {
        int n = nt * 16 + ln;
        bias[nt] = b_e[n]; gg[nt] = g_e[n]; bl[nt] = b_ln[n];
    }
    #pragma unroll
    for (int r = 0; r < 8; ++r) {
        float v[8];
        float s = 0.0f, ss = 0.0f;
        #pragma unroll
        for (int nt = 0; nt < 8; ++nt) {
            v[nt] = acc[nt][r] + bias[nt];
            s += v[nt]; ss += v[nt] * v[nt];
        }
        // butterfly within each 16-lane half (rows r / r+8 stay separate)
        #pragma unroll
        for (int m = 1; m <= 8; m <<= 1) {
            s  += __shfl_xor(s,  m, 32);
            ss += __shfl_xor(ss, m, 32);
        }
        float mu  = s * (1.0f / 128.0f);
        float var = ss * (1.0f / 128.0f) - mu * mu;
        float rs  = rsqrtf(var + 1e-5f);
        int row = r + hi * 8;
        size_t base = (size_t)(e0 + row) * 128 + ln;
        #pragma unroll
        for (int nt = 0; nt < 8; ++nt)
            out[base + nt * 16] = (v[nt] - mu) * rs * gg[nt] + bl[nt];
    }
}

extern "C" void kernel_launch(void* const* d_in, const int* in_sizes, int n_in,
                              void* d_out, int out_size, void* d_ws, size_t ws_size,
                              hipStream_t stream) {
    const float* X      = (const float*)d_in[0];
    const float* mask   = (const float*)d_in[1];
    const int*   residx = (const int*)d_in[2];
    const int*   chains = (const int*)d_in[3];
    const float* W_e    = (const float*)d_in[4];
    const float* b_e    = (const float*)d_in[5];
    const float* g_e    = (const float*)d_in[6];
    const float* be_ln  = (const float*)d_in[7];
    const float* W_pe   = (const float*)d_in[8];
    const float* b_pe   = (const float*)d_in[9];

    float* out = (float*)d_out;
    float*     Cb = (float*)d_ws;                               // 16384*3*4   = 196608 B
    _Float16*  Wt = (_Float16*)((char*)d_ws + 196608);          // 128*416*2   = 106496 B
    int*       Eidx = (int*)(out + PF_EOUT);                    // tail of d_out (int bits)

    pf_cb_kernel<<<64, 256, 0, stream>>>(X, Cb, PF_B * PF_L);
    pf_wt_kernel<<<208, 256, 0, stream>>>(W_e, Wt, 128 * PF_KDIM);
    pf_topk_kernel<<<PF_B * PF_L, 256, 0, stream>>>(X, mask, Eidx);
    pf_edge_gemm_kernel<<<(PF_EDGES / 16) / PF_WAVES, 128, 0, stream>>>(
        X, Cb, residx, chains, Wt, b_e, g_e, be_ln, W_pe, b_pe, Eidx, out);
}
